// MSDeformableAttention_77816217469504
// MI455X (gfx1250) — matrix operand order repaired and checked by reading, not compiled, not run on hardware
//
#include <hip/hip_runtime.h>
#include <hip/hip_bf16.h>

typedef _Float16 v16h __attribute__((ext_vector_type(16)));
typedef float    v8f  __attribute__((ext_vector_type(8)));
typedef float    v4f  __attribute__((ext_vector_type(4)));

namespace cfg {
constexpr int N_HEADQ  = 8;
constexpr int N_LEVELS = 4;
constexpr int N_POINTS = 4;
constexpr int EMBED    = 256;
constexpr int HEAD_DIM = 32;
constexpr int BS       = 2;
constexpr int LQ       = 13294;           // == LEN_V
constexpr int LEN_V    = 13294;
constexpr int MROWS    = BS * LQ;         // 26588
constexpr int KTOT     = 256;             // K of every GEMM in this op
constexpr int LDS_STRIDE = 17;            // 16 cols + 1 pad dword (bank-conflict free)
}

// ---------------------------------------------------------------------------
// WMMA GEMM: Y[M,N] = X[M,256] @ W[256,N] + bias[N]  (fp32 in/out, f16 WMMA)
// Block = 128 threads = 4 waves; each wave owns one 16x16 output tile
// (block tile = 64 rows x 16 cols). grid.x = N/16, grid.y = ceil(M/64).
//
// Data movement:
//   * W panel [256 x 16] is staged into LDS once per block via
//     global_load_async_to_lds_b128 (ASYNCcnt) -> 4x reuse across waves.
//   * A fragments load as 2x32B float4 pairs (global_load_b128).
// Fragment layouts per CDNA5 ISA 7.12.2 (wave32):
//   A (16x32 f16):  lane l: row = l&15; kb = 8*(l>>4);
//                   a[i] = X[row, kt + kb + i + (i>=8 ? 8 : 0)]
//   B (32x16 f16):  lane l: col = l&15; kb = 8*(l>>4); b[i] = W[kt+kb+i+(i>=8 ? 8 : 0), col]
//   C/D (16x16 f32): vgpr r: row = r + 8*(l>>4); col = l&15
// ---------------------------------------------------------------------------
__global__ __launch_bounds__(128) void gemm_wmma_f16(
    const float* __restrict__ X, const float* __restrict__ W,
    const float* __restrict__ bias, float* __restrict__ Y,
    int M, int N)
{
    using namespace cfg;
    __shared__ float sB[KTOT * LDS_STRIDE];   // 17 KB staged W panel (fp32)

    const int t     = threadIdx.x;            // 0..127
    const int lane  = t & 31;
    const int wave  = t >> 5;
    const int hsel  = lane >> 4;              // 0 or 1
    const int lrow  = lane & 15;

    const int tileM = blockIdx.y * 64 + wave * 16;
    const int tileN = blockIdx.x * 16;

    // ---- Stage W[0:256, tileN:tileN+16] into LDS via async-to-LDS DMA ----
    {
        const unsigned ldsbase = (unsigned)(size_t)&sB[0];
        const int krow = t >> 2;              // 0..31
        const int cg   = (t & 3) * 4;         // col group: 0,4,8,12
#pragma unroll
        for (int it = 0; it < 8; ++it) {
            const int k = it * 32 + krow;
            const float* gsrc = W + (unsigned)k * (unsigned)N + (unsigned)(tileN + cg);
            const unsigned ldst = ldsbase + ((unsigned)k * LDS_STRIDE + (unsigned)cg) * 4u;
            asm volatile("global_load_async_to_lds_b128 %0, %1, off"
                         :: "v"(ldst), "v"(gsrc)
                         : "memory");
        }
        asm volatile("s_wait_asynccnt 0x0" ::: "memory");
    }
    __syncthreads();

    // ---- A row pointer (reads clamped; writes guarded) ----
    int aRow = tileM + lrow;
    if (aRow > M - 1) aRow = M - 1;
    const float* __restrict__ Arow = X + (unsigned)aRow * (unsigned)KTOT;

    const int col = tileN + lrow;
    const int kbA = 8 * hsel;
    const int kbB = 8 * hsel;

    v8f acc = {};

#pragma unroll
    for (int kt = 0; kt < KTOT; kt += 32) {
        if (kt + 32 < KTOT)
            __builtin_prefetch(Arow + kt + 32, 0, 1);   // global_prefetch_b8

        // A fragment: two aligned 32B chunks (k: kbA..+7 and kbA+16..+23)
        const v4f* ap0 = (const v4f*)(Arow + kt + kbA);
        const v4f* ap1 = (const v4f*)(Arow + kt + kbA + 16);
        const v4f a0 = ap0[0], a1 = ap0[1];
        const v4f a2 = ap1[0], a3 = ap1[1];

        v16h a, b;
#pragma unroll
        for (int j = 0; j < 4; ++j) {
            a[j]      = (_Float16)a0[j];
            a[4 + j]  = (_Float16)a1[j];
            a[8 + j]  = (_Float16)a2[j];
            a[12 + j] = (_Float16)a3[j];
        }
        // B fragment from LDS (ds_load_b32, conflict-free via padded stride)
#pragma unroll
        for (int i = 0; i < 16; ++i) {
            b[i] = (_Float16)sB[(kt + kbB + i + (i >= 8 ? 8 : 0)) * LDS_STRIDE + lrow];
        }
        // v_wmma_f32_16x16x32_f16
        acc = __builtin_amdgcn_wmma_f32_16x16x32_f16(
            /*neg_a=*/false, a, /*neg_b=*/false, b,
            /*c_mod=*/(short)0, acc, /*reuse_a=*/false, /*reuse_b=*/false);
    }

    // ---- Epilogue: 32-bit offsets; wave-uniform fast path ----
    const float bv = bias[col];
    const unsigned ybase = (unsigned)tileM * (unsigned)N + (unsigned)col;
    if (tileM + 16 <= M) {
#pragma unroll
        for (int r = 0; r < 8; ++r)
            Y[ybase + (unsigned)(r + 8 * hsel) * (unsigned)N] = acc[r] + bv;
    } else {
#pragma unroll
        for (int r = 0; r < 8; ++r) {
            const int row = tileM + r + 8 * hsel;
            if (row < M)
                Y[(unsigned)row * (unsigned)N + (unsigned)col] = acc[r] + bv;
        }
    }
}

// ---------------------------------------------------------------------------
// Softmax over the trailing 16 (levels*points) per (b,q,head), in place.
// ---------------------------------------------------------------------------
__global__ void softmax16_kernel(float* __restrict__ a, int n)
{
    const int i = blockIdx.x * blockDim.x + threadIdx.x;
    if (i >= n) return;
    float* p = a + (size_t)i * 16;
    float m = p[0];
#pragma unroll
    for (int j = 1; j < 16; ++j) m = fmaxf(m, p[j]);
    float e[16];
    float s = 0.f;
#pragma unroll
    for (int j = 0; j < 16; ++j) { e[j] = expf(p[j] - m); s += e[j]; }
    const float inv = 1.f / s;
#pragma unroll
    for (int j = 0; j < 16; ++j) p[j] = e[j] * inv;
}

// ---------------------------------------------------------------------------
// Bilinear sampling + attention aggregation.
// grid = (LQ, BS); block = 256 threads = 8 waves; wave = head, lane = channel.
// vproj layout [b, Lv, head, 32] -> lane gather is one contiguous 128B line.
// Invalid corners are skipped (reference zeroes their weights).
// ---------------------------------------------------------------------------
__global__ __launch_bounds__(256) void sample_agg_kernel(
    const float* __restrict__ vproj, const float* __restrict__ off,
    const float* __restrict__ attn,  const float* __restrict__ refp,
    float* __restrict__ agg)
{
    using namespace cfg;
    const int LH[4] = {100, 50, 25, 13};
    const int LW[4] = {100, 50, 25, 13};
    const int LSTART[4] = {0, 10000, 12500, 13125};

    const int q    = blockIdx.x;
    const int b    = blockIdx.y;
    const int head = threadIdx.x >> 5;
    const int c    = threadIdx.x & 31;

    const size_t bq   = (size_t)b * LQ + q;
    const float* offp = off  + (bq * N_HEADQ + head) * (N_LEVELS * N_POINTS * 2);
    const float* awp  = attn + (bq * N_HEADQ + head) * (N_LEVELS * N_POINTS);
    const float* vb   = vproj + ((size_t)b * LEN_V) * EMBED + head * HEAD_DIM + c;

    float acc = 0.f;

#pragma unroll
    for (int lvl = 0; lvl < N_LEVELS; ++lvl) {
        const int   h = LH[lvl], w = LW[lvl];
        const float fw = (float)w, fh = (float)h;
        const float rx = refp[(bq * N_LEVELS + lvl) * 2 + 0];
        const float ry = refp[(bq * N_LEVELS + lvl) * 2 + 1];
        const float* vl = vb + (size_t)LSTART[lvl] * EMBED;

#pragma unroll
        for (int p = 0; p < N_POINTS; ++p) {
            const float ox = offp[(lvl * N_POINTS + p) * 2 + 0];
            const float oy = offp[(lvl * N_POINTS + p) * 2 + 1];
            // loc = ref + off/normalizer(w,h); pixel coords (align_corners=False):
            const float x = (rx + ox / fw) * fw - 0.5f;
            const float y = (ry + oy / fh) * fh - 0.5f;
            const float x0f = floorf(x), y0f = floorf(y);
            const int   x0 = (int)x0f,  y0 = (int)y0f;
            const float wx1 = x - x0f,  wy1 = y - y0f;
            const float wx0 = 1.f - wx1, wy0 = 1.f - wy1;
            const float aw = awp[lvl * N_POINTS + p];

            float s = 0.f;
#pragma unroll
            for (int dy = 0; dy < 2; ++dy) {
#pragma unroll
                for (int dx = 0; dx < 2; ++dx) {
                    const int xi = x0 + dx, yi = y0 + dy;
                    if (xi >= 0 && xi < w && yi >= 0 && yi < h) {
                        const float wt = (dx ? wx1 : wx0) * (dy ? wy1 : wy0);
                        s += wt * vl[(size_t)(yi * w + xi) * EMBED];
                    }
                }
            }
            acc += aw * s;
        }
    }

    agg[bq * EMBED + head * HEAD_DIM + c] = acc;
}

// ---------------------------------------------------------------------------
extern "C" void kernel_launch(void* const* d_in, const int* in_sizes, int n_in,
                              void* d_out, int out_size, void* d_ws, size_t ws_size,
                              hipStream_t stream)
{
    using namespace cfg;
    (void)in_sizes; (void)n_in; (void)out_size; (void)ws_size;

    const float* query  = (const float*)d_in[0];
    const float* refp   = (const float*)d_in[1];
    const float* value  = (const float*)d_in[2];
    /* d_in[3] = value_spatial_shapes (static; hardcoded) */
    const float* W_off  = (const float*)d_in[4];
    const float* b_off  = (const float*)d_in[5];
    const float* W_attn = (const float*)d_in[6];
    const float* b_attn = (const float*)d_in[7];
    const float* W_val  = (const float*)d_in[8];
    const float* b_val  = (const float*)d_in[9];
    const float* W_out  = (const float*)d_in[10];
    const float* b_out  = (const float*)d_in[11];
    float* out = (float*)d_out;

    // Workspace partition (floats): vproj | off | attn | agg  (~95.3 MB total)
    float* vproj = (float*)d_ws;
    float* offb  = vproj + (size_t)MROWS * EMBED;
    float* attnb = offb  + (size_t)MROWS * EMBED;
    float* aggb  = attnb + (size_t)MROWS * 128;

    const dim3 blk(128);
    const dim3 gN256(EMBED / 16, (MROWS + 63) / 64);
    const dim3 gN128(128 / 16,  (MROWS + 63) / 64);

    // 1) value projection: [MROWS,256] @ [256,256]
    gemm_wmma_f16<<<gN256, blk, 0, stream>>>(value, W_val, b_val, vproj,
                                             MROWS, EMBED);
    // 2) sampling offsets: [MROWS,256] @ [256,256]
    gemm_wmma_f16<<<gN256, blk, 0, stream>>>(query, W_off, b_off, offb,
                                             MROWS, EMBED);
    // 3) attention logits: [MROWS,256] @ [256,128]
    gemm_wmma_f16<<<gN128, blk, 0, stream>>>(query, W_attn, b_attn, attnb,
                                             MROWS, 128);
    // 4) softmax over 16 per (b,q,head)
    {
        const int n = MROWS * N_HEADQ;
        softmax16_kernel<<<(n + 255) / 256, 256, 0, stream>>>(attnb, n);
    }
    // 5) bilinear sample + aggregate -> agg[b,q,head,32]
    sample_agg_kernel<<<dim3(LQ, BS), 256, 0, stream>>>(vproj, offb, attnb,
                                                        refp, aggb);
    // 6) output projection -> d_out
    gemm_wmma_f16<<<gN256, blk, 0, stream>>>(aggb, W_out, b_out, out,
                                             MROWS, EMBED);
}
